// GATModel_90383291777330
// MI455X (gfx1250) — matrix operand order repaired and checked by reading, not compiled, not run on hardware
//
#include <hip/hip_runtime.h>
#include <math.h>

#define NN    50000
#define EE    800000
#define TE    (EE + NN)      // edges + self loops = 850000
#define FIN   128
#define HIDC  128
#define NH1   4
#define NG    512
#define NEG_SLOPE 0.2f

typedef __attribute__((ext_vector_type(16))) _Float16 v16h;
typedef __attribute__((ext_vector_type(8)))  float    v8f;

union FragU { v16h h; unsigned u[8]; };

__device__ __forceinline__ int   fkey(float f) { int b = __float_as_int(f); return b >= 0 ? b : (b ^ 0x7fffffff); }
__device__ __forceinline__ float fdec(int k)   { return __int_as_float(k >= 0 ? k : (k ^ 0x7fffffff)); }
__device__ __forceinline__ float lrelu(float x){ return x > 0.f ? x : NEG_SLOPE * x; }
__device__ __forceinline__ float eluf(float x) { return x > 0.f ? x : (expf(x) - 1.f); }

// ---------------- init / conversion ----------------

__global__ void k_fill_i32(int* p, int n, int v) {
    int t = blockIdx.x * blockDim.x + threadIdx.x;
    if (t < n) p[t] = v;
}

__global__ void k_f32_to_f16(const float* __restrict__ a, _Float16* __restrict__ b, long n) {
    long t = (long)blockIdx.x * blockDim.x + threadIdx.x;
    if (t < n) b[t] = (_Float16)a[t];
}

// W[k, c] (row-major KxNcol) -> Wt[c, k] f16 (column-major: K contiguous per output col)
__global__ void k_w_T(const float* __restrict__ W, _Float16* __restrict__ Wt, int K, int Ncol) {
    long t = (long)blockIdx.x * blockDim.x + threadIdx.x;
    if (t >= (long)K * Ncol) return;
    int k = (int)(t / Ncol), c = (int)(t % Ncol);
    Wt[(long)c * K + k] = (_Float16)W[t];
}

// ---------------- WMMA GEMM with async-to-LDS A staging ----------------
// Block = 8 waves. NTG col-groups of 64; rpb = 8/NTG row strips per block.
// A strips staged in LDS via global_load_async_to_lds_b128 (shared by waves);
// B (weights, L2-resident) streamed as register fragments. K compile-time ->
// fully unrolled loop, no manual double-buffer (avoids spills seen in round 2).

template <int K, int NTG>
__global__ void k_gemm_wmma(const _Float16* __restrict__ A, const _Float16* __restrict__ Bt,
                            float* __restrict__ C, int M) {
    constexpr int RPB  = 8 / NTG;
    constexpr int NCOL = NTG * 64;
    extern __shared__ _Float16 sA[];           // RPB*16 rows x K, row-major
    int tid  = threadIdx.x;
    int wave = tid >> 5, lane = tid & 31;

    // ---- async stage A strips into LDS (CDNA5 async path, ASYNCcnt) ----
    long rowbase  = (long)blockIdx.x * (RPB * 16);
    long rows_rem = (long)M - rowbase;
    int  rows_here = rows_rem < (long)(RPB * 16) ? (int)rows_rem : (RPB * 16);
    int  bytes_valid = rows_here * K * 2;
    {
        const char* gsrc = (const char*)(A + rowbase * (long)K);
        unsigned lbase = (unsigned)(unsigned long long)sA;
        for (int off = tid * 16; off < bytes_valid; off += 256 * 16) {
            unsigned laddr = lbase + off;
            const char* ga = gsrc + off;
            asm volatile("global_load_async_to_lds_b128 %0, %1, off"
                         :: "v"(laddr), "v"(ga) : "memory");
        }
        asm volatile("s_wait_asynccnt 0" ::: "memory");
    }
    __syncthreads();

    int strip = wave / NTG;
    int cg    = wave - strip * NTG;
    long crow0 = rowbase + strip * 16;
    if (crow0 + 16 <= (long)M) {               // wave-uniform: EXEC all-ones for WMMA
        int half = lane >> 4, m15 = lane & 15;
        const _Float16* sArow = sA + (long)(strip * 16 + m15) * K;
        int colbase = cg * 64;
        // per-j B base pointers (K contiguous per column in Bt)
        const _Float16* Bb0 = Bt + (long)(colbase +  0 + m15) * K + half * 16;
        const _Float16* Bb1 = Bt + (long)(colbase + 16 + m15) * K + half * 16;
        const _Float16* Bb2 = Bt + (long)(colbase + 32 + m15) * K + half * 16;
        const _Float16* Bb3 = Bt + (long)(colbase + 48 + m15) * K + half * 16;

        v8f acc[4] = {};
        #pragma unroll
        for (int k0 = 0; k0 < K; k0 += 32) {
            FragU a;
            #pragma unroll
            for (int i = 0; i < 8; ++i) {
                int kk = half * 16 + 2 * i;
                a.u[i] = *reinterpret_cast<const unsigned*>(sArow + k0 + kk);  // ds_load
            }
            FragU b[4];
            #pragma unroll
            for (int i = 0; i < 8; ++i) b[0].u[i] = *reinterpret_cast<const unsigned*>(Bb0 + k0 + 2 * i);
            #pragma unroll
            for (int i = 0; i < 8; ++i) b[1].u[i] = *reinterpret_cast<const unsigned*>(Bb1 + k0 + 2 * i);
            #pragma unroll
            for (int i = 0; i < 8; ++i) b[2].u[i] = *reinterpret_cast<const unsigned*>(Bb2 + k0 + 2 * i);
            #pragma unroll
            for (int i = 0; i < 8; ++i) b[3].u[i] = *reinterpret_cast<const unsigned*>(Bb3 + k0 + 2 * i);
            #pragma unroll
            for (int j = 0; j < 4; ++j)
                acc[j] = __builtin_amdgcn_wmma_f32_16x16x32_f16(
                    false, a.h, false, b[j].h, (short)0, acc[j], false, false);
        }

        int rbase = (int)crow0 + ((lane >> 4) << 3);   // lanes 16..31: rows M+8..M+15
        #pragma unroll
        for (int j = 0; j < 4; ++j) {
            int col = colbase + j * 16 + m15;
            #pragma unroll
            for (int r = 0; r < 8; ++r)
                C[(long)(rbase + r) * NCOL + col] = acc[j][r];
        }
    }
}

// ---------------- attention coefficients: asrc[n,h] = <h[n,h,:], a_src[h,:]> ----------------

__global__ void k_attn(const float* __restrict__ h, const float* __restrict__ a_src,
                       const float* __restrict__ a_dst, float* __restrict__ asrc,
                       float* __restrict__ adst, int H) {
    int t = blockIdx.x * blockDim.x + threadIdx.x;
    if (t >= NN * H) return;
    int n = t / H, hd = t - n * H;
    const float* hp = h + (long)n * (H * HIDC) + hd * HIDC;
    const float* as = a_src + hd * HIDC;
    const float* ad = a_dst + hd * HIDC;
    float s1 = 0.f, s2 = 0.f;
    #pragma unroll 8
    for (int c = 0; c < HIDC; ++c) { float v = hp[c]; s1 += v * as[c]; s2 += v * ad[c]; }
    asrc[t] = s1; adst[t] = s2;
}

// ---------------- edge passes (segment max / exp-sum / weighted aggregate) ----------------

__device__ __forceinline__ void edge_pair(const int* __restrict__ ei, int e, int& s, int& d) {
    if (e < EE) { s = ei[e]; d = ei[EE + e]; } else { s = d = e - EE; }  // self loops appended
}

__global__ void k_edge_max(const int* __restrict__ ei, const float* __restrict__ asrc,
                           const float* __restrict__ adst, int* __restrict__ mkey, int H) {
    int t = blockIdx.x * blockDim.x + threadIdx.x;
    if (t >= TE * H) return;
    int e = t / H, hd = t - e * H;
    int s, d; edge_pair(ei, e, s, d);
    float lg = lrelu(asrc[s * H + hd] + adst[d * H + hd]);
    atomicMax(mkey + d * H + hd, fkey(lg));
}

__global__ void k_edge_expsum(const int* __restrict__ ei, const float* __restrict__ asrc,
                              const float* __restrict__ adst, const int* __restrict__ mkey,
                              float* __restrict__ denom, int H) {
    int t = blockIdx.x * blockDim.x + threadIdx.x;
    if (t >= TE * H) return;
    int e = t / H, hd = t - e * H;
    int s, d; edge_pair(ei, e, s, d);
    float lg = lrelu(asrc[s * H + hd] + adst[d * H + hd]);
    atomicAdd(denom + d * H + hd, expf(lg - fdec(mkey[d * H + hd])));
}

// one block (128 threads = channels) per edge; loops heads
__global__ void k_edge_agg(const int* __restrict__ ei, const float* __restrict__ asrc,
                           const float* __restrict__ adst, const int* __restrict__ mkey,
                           const float* __restrict__ denom, const float* __restrict__ h,
                           float* __restrict__ out, int H) {
    int e = blockIdx.x, c = threadIdx.x;
    int s, d; edge_pair(ei, e, s, d);
    for (int hd = 0; hd < H; ++hd) {
        float lg    = lrelu(asrc[s * H + hd] + adst[d * H + hd]);
        float alpha = expf(lg - fdec(mkey[d * H + hd])) / denom[d * H + hd];
        atomicAdd(out + (long)d * (H * HIDC) + hd * HIDC + c,
                  alpha * h[(long)s * (H * HIDC) + hd * HIDC + c]);
    }
}

// ---------------- epilogues ----------------

__global__ void k_bias_elu_h(const float* __restrict__ agg, const float* __restrict__ bias,
                             _Float16* __restrict__ out, long total, int ncols) {
    long t = (long)blockIdx.x * blockDim.x + threadIdx.x;
    if (t >= total) return;
    int c = (int)(t % ncols);
    out[t] = (_Float16)eluf(agg[t] + bias[c]);
}

__global__ void k_pool(const float* __restrict__ h2agg, const float* __restrict__ b2,
                       const int* __restrict__ batch, float* __restrict__ sums,
                       float* __restrict__ cnts) {
    long t = (long)blockIdx.x * blockDim.x + threadIdx.x;
    if (t >= (long)NN * HIDC) return;
    int n = (int)(t >> 7), c = (int)(t & 127);
    float v = eluf(h2agg[t] + b2[c]);
    int g = batch[n];
    atomicAdd(sums + g * HIDC + c, v);
    if (c == 0) atomicAdd(cnts + g, 1.0f);
}

__global__ void k_final(const float* __restrict__ sums, const float* __restrict__ cnts,
                        const float* __restrict__ Wl, const float* __restrict__ bl,
                        float* __restrict__ out) {
    __shared__ float red[HIDC];
    int g = blockIdx.x, c = threadIdx.x;
    float cnt = fmaxf(cnts[g], 1.0f);
    red[c] = (sums[g * HIDC + c] / cnt) * Wl[c];
    __syncthreads();
    for (int s = HIDC / 2; s > 0; s >>= 1) { if (c < s) red[c] += red[c + s]; __syncthreads(); }
    if (c == 0) out[g] = red[0] + bl[0];
}

// ---------------- host ----------------

extern "C" void kernel_launch(void* const* d_in, const int* in_sizes, int n_in,
                              void* d_out, int out_size, void* d_ws, size_t ws_size,
                              hipStream_t stream) {
    (void)in_sizes; (void)n_in; (void)out_size; (void)ws_size;
    const float* x      = (const float*)d_in[0];
    const int*   ei     = (const int*)d_in[1];
    const int*   batch  = (const int*)d_in[2];
    const float* W1     = (const float*)d_in[3];
    const float* a_src1 = (const float*)d_in[4];
    const float* a_dst1 = (const float*)d_in[5];
    const float* b1     = (const float*)d_in[6];
    const float* W2     = (const float*)d_in[7];
    const float* a_src2 = (const float*)d_in[8];
    const float* a_dst2 = (const float*)d_in[9];
    const float* b2     = (const float*)d_in[10];
    const float* Wl     = (const float*)d_in[11];
    const float* bl     = (const float*)d_in[12];
    float* out = (float*)d_out;

    char* base = (char*)d_ws; size_t off = 0;
    auto alloc = [&](size_t bytes) -> void* {
        void* p = base + off; off = (off + bytes + 255) & ~(size_t)255; return p;
    };
    _Float16* xh    = (_Float16*)alloc((size_t)NN * FIN * 2);
    _Float16* wt1   = (_Float16*)alloc((size_t)FIN * (NH1 * HIDC) * 2);
    _Float16* wt2   = (_Float16*)alloc((size_t)(NH1 * HIDC) * HIDC * 2);
    float*    h1    = (float*)   alloc((size_t)NN * (NH1 * HIDC) * 4);
    _Float16* h1h   = (_Float16*)alloc((size_t)NN * (NH1 * HIDC) * 2);
    float*    h2    = (float*)   alloc((size_t)NN * HIDC * 4);
    float*    asrc1 = (float*)   alloc((size_t)NN * NH1 * 4);
    float*    adst1 = (float*)   alloc((size_t)NN * NH1 * 4);
    float*    asrc2 = (float*)   alloc((size_t)NN * 4);
    float*    adst2 = (float*)   alloc((size_t)NN * 4);
    int*      mkey1 = (int*)     alloc((size_t)NN * NH1 * 4);
    int*      mkey2 = (int*)     alloc((size_t)NN * 4);
    size_t zoff = off;
    float*    h1agg  = (float*)alloc((size_t)NN * (NH1 * HIDC) * 4);
    float*    h2agg  = (float*)alloc((size_t)NN * HIDC * 4);
    float*    denom1 = (float*)alloc((size_t)NN * NH1 * 4);
    float*    denom2 = (float*)alloc((size_t)NN * 4);
    float*    sums   = (float*)alloc((size_t)NG * HIDC * 4);
    float*    cnts   = (float*)alloc((size_t)NG * 4);
    size_t zbytes = off - zoff;

    // self-initializing every call (graph-replay safe)
    hipMemsetAsync(base + zoff, 0, zbytes, stream);
    k_fill_i32<<<(NN * NH1 + 255) / 256, 256, 0, stream>>>(mkey1, NN * NH1, (int)0x80000000);
    k_fill_i32<<<(NN + 255) / 256, 256, 0, stream>>>(mkey2, NN, (int)0x80000000);

    long nx = (long)NN * FIN;
    k_f32_to_f16<<<(int)((nx + 255) / 256), 256, 0, stream>>>(x, xh, nx);
    k_w_T<<<(FIN * NH1 * HIDC + 255) / 256, 256, 0, stream>>>(W1, wt1, FIN, NH1 * HIDC);
    k_w_T<<<(NH1 * HIDC * HIDC + 255) / 256, 256, 0, stream>>>(W2, wt2, NH1 * HIDC, HIDC);

    // ---- layer 1: GEMM  K=128, NTG=8 (rpb=1, LDS stage = 16*128*2 = 4KB) ----
    {   int strips = NN / 16, rpb = 1;
        int blocks = (strips + rpb - 1) / rpb;
        size_t lds = (size_t)rpb * 16 * FIN * 2;
        k_gemm_wmma<FIN, 8><<<blocks, 256, lds, stream>>>(xh, wt1, h1, NN); }
    k_attn<<<(NN * NH1 + 255) / 256, 256, 0, stream>>>(h1, a_src1, a_dst1, asrc1, adst1, NH1);
    k_edge_max   <<<(TE * NH1 + 255) / 256, 256, 0, stream>>>(ei, asrc1, adst1, mkey1, NH1);
    k_edge_expsum<<<(TE * NH1 + 255) / 256, 256, 0, stream>>>(ei, asrc1, adst1, mkey1, denom1, NH1);
    k_edge_agg   <<<TE, HIDC, 0, stream>>>(ei, asrc1, adst1, mkey1, denom1, h1, h1agg, NH1);
    long n1 = (long)NN * (NH1 * HIDC);
    k_bias_elu_h<<<(int)((n1 + 255) / 256), 256, 0, stream>>>(h1agg, b1, h1h, n1, NH1 * HIDC);

    // ---- layer 2: GEMM  K=512, NTG=2 (rpb=4, LDS stage = 64*512*2 = 64KB) ----
    {   int strips = NN / 16, rpb = 4;
        int blocks = (strips + rpb - 1) / rpb;
        size_t lds = (size_t)rpb * 16 * (NH1 * HIDC) * 2;
        k_gemm_wmma<NH1 * HIDC, 2><<<blocks, 256, lds, stream>>>(h1h, wt2, h2, NN); }
    k_attn<<<(NN + 255) / 256, 256, 0, stream>>>(h2, a_src2, a_dst2, asrc2, adst2, 1);
    k_edge_max   <<<(TE + 255) / 256, 256, 0, stream>>>(ei, asrc2, adst2, mkey2, 1);
    k_edge_expsum<<<(TE + 255) / 256, 256, 0, stream>>>(ei, asrc2, adst2, mkey2, denom2, 1);
    k_edge_agg   <<<TE, HIDC, 0, stream>>>(ei, asrc2, adst2, mkey2, denom2, h2, h2agg, 1);

    // ---- pool + head ----
    long n2 = (long)NN * HIDC;
    k_pool<<<(int)((n2 + 255) / 256), 256, 0, stream>>>(h2agg, b2, batch, sums, cnts);
    k_final<<<NG, HIDC, 0, stream>>>(sums, cnts, Wl, bl, out);
}
